// BioLatentMoELayer_83906481095352
// MI455X (gfx1250) — matrix operand order repaired and checked by reading, not compiled, not run on hardware
//
#include <hip/hip_runtime.h>

typedef unsigned short u16;
typedef unsigned int   u32;
typedef __attribute__((ext_vector_type(16))) __bf16 v16bf;
typedef __attribute__((ext_vector_type(8)))  float  v8f;

// Problem constants (from reference)
constexpr int N_  = 2048 * 4;   // S*B tokens
constexpr int D_  = 2048;
constexpr int L_  = 256;
constexpr int E_  = 32;
constexpr int EH_ = 1024;
constexpr int SH_ = 2048;
constexpr float AUX_ = 1e-4f;

__device__ __forceinline__ u16 f2bf(float f) {
  u32 u = __float_as_uint(f);
  u32 r = 0x7FFFu + ((u >> 16) & 1u);   // round-to-nearest-even
  return (u16)((u + r) >> 16);
}
__device__ __forceinline__ float bf2f(u16 b) {
  return __uint_as_float(((u32)b) << 16);
}

// ---------------------------------------------------------------------------
// Tiled bf16 WMMA GEMM: C(MxN) = A(MxK) @ B(KxN), fp32 accumulate.
// B is supplied TRANSPOSED (BT[N][K], ldbt==K) so both A and B WMMA fragments
// are contiguous-in-K and load from LDS as ds_load_b128 pairs.
// Block tile 128x128, K-step 32, double-buffered LDS (one barrier / K-step).
// 8 waves in a 4(M)x2(N) grid; each wave owns a 32x64 patch = 2x4
// v_wmma_f32_16x16x32_bf16 accumulators.
// Out-of-range A rows are CLAMPED (not zero-filled): row m only affects C row
// m, and rows >= Meff are masked in the epilogue, so staging stays branchless.
// ---------------------------------------------------------------------------
#define BM 128
#define BN 128
#define BK 32

enum { EPI_STORE = 0, EPI_SCATTER_SCALE_ACC = 1, EPI_GATE_ADD = 2, EPI_RESID = 3 };

__global__ __launch_bounds__(256)
void gemm_bf16_wmma(const u16* __restrict__ A, int lda,
                    const u16* __restrict__ BT, int ldbt,   // BT[N][K], ldbt = K
                    int M, int Ncols, int K,
                    const int* __restrict__ rowidx,     // A-row gather list (or null)
                    const int* __restrict__ cntp,       // device row count (or null)
                    const int* __restrict__ scatteridx, // epilogue scatter list (mode 1)
                    float* __restrict__ C, u16* __restrict__ Cbf, int ldc,
                    int mode,
                    const float* __restrict__ comb, int expert, int nexp,
                    float* __restrict__ Cacc,
                    const float* __restrict__ gatev,
                    const float* __restrict__ addend,
                    const float* __restrict__ resid)
{
  __shared__ __align__(16) u16 As[2][BM][BK];   // 2 x 8 KB
  __shared__ __align__(16) u16 Bs[2][BN][BK];   // 2 x 8 KB

  const int Meff = cntp ? *cntp : M;
  const int row0 = blockIdx.x * BM;
  if (row0 >= Meff) return;                 // block-uniform exit (EXEC stays full)
  const int col0 = blockIdx.y * BN;

  const int tid  = threadIdx.x;
  const int lane = tid & 31;
  const int wid  = tid >> 5;
  const int wm   = wid & 3;                 // wave row (4) -> 32 rows each
  const int wn   = wid >> 2;                // wave col (2) -> 64 cols each

  const v8f vzero = {0.f,0.f,0.f,0.f,0.f,0.f,0.f,0.f};
  v8f acc[2][4];
#pragma unroll
  for (int sm = 0; sm < 2; ++sm)
#pragma unroll
    for (int sn = 0; sn < 4; ++sn) acc[sm][sn] = vzero;

  // ---- staging assignments: 2 uint4 chunks per thread for each tile ----
  const int tr0 = tid >> 2;                 // 0..63
  const int tr1 = tr0 + 64;                 // 64..127
  const int tsg = (tid & 3) << 3;           // 0,8,16,24 (8 bf16 per chunk)

  // Clamped A-row gather (branchless staging; masked rows handled in epilogue).
  const int last = Meff - 1;
  int g0 = row0 + tr0; if (g0 > last) g0 = last;
  int g1 = row0 + tr1; if (g1 > last) g1 = last;
  const int sr0 = rowidx ? rowidx[g0] : g0;
  const int sr1 = rowidx ? rowidx[g1] : g1;
  const u16* ap0 = A + (size_t)sr0 * lda + tsg;
  const u16* ap1 = A + (size_t)sr1 * lda + tsg;
  const u16* bp0 = BT + (size_t)(col0 + tr0) * ldbt + tsg;
  const u16* bp1 = BT + (size_t)(col0 + tr1) * ldbt + tsg;

  // fragment addressing (contiguous-K order, same for A and B)
  const int hi  = lane >> 4;
  const int l16 = lane & 15;
  const int ak  = hi << 4;    // A: lanes<16 K0..15; lanes>=16 K16..31
  const int bk  = hi << 4;    // B: lanes<16 K0..15; lanes>=16 K16..31

  union Frag { u16 u[16]; uint4 q[2]; v16bf v; };

  const int nsteps = K / BK;

  // prologue: stage tile 0 into buffer 0
  {
    *(uint4*)&As[0][tr0][tsg] = *(const uint4*)(ap0);
    *(uint4*)&As[0][tr1][tsg] = *(const uint4*)(ap1);
    *(uint4*)&Bs[0][tr0][tsg] = *(const uint4*)(bp0);
    *(uint4*)&Bs[0][tr1][tsg] = *(const uint4*)(bp1);
  }
  __syncthreads();

  for (int s = 0; s < nsteps; ++s) {
    const int cur = s & 1;
    const int nxt = cur ^ 1;

    // stage next tile while computing current one
    if (s + 1 < nsteps) {
      const int kn = (s + 1) * BK;
      *(uint4*)&As[nxt][tr0][tsg] = *(const uint4*)(ap0 + kn);
      *(uint4*)&As[nxt][tr1][tsg] = *(const uint4*)(ap1 + kn);
      *(uint4*)&Bs[nxt][tr0][tsg] = *(const uint4*)(bp0 + kn);
      *(uint4*)&Bs[nxt][tr1][tsg] = *(const uint4*)(bp1 + kn);
      if (s + 2 < nsteps) {   // prefetch tile s+2 -> global_prefetch_b8
        __builtin_prefetch(bp0 + kn + BK, 0, 1);
        __builtin_prefetch(bp1 + kn + BK, 0, 1);
      }
    }

    Frag af[2], bfr[4];
#pragma unroll
    for (int sm = 0; sm < 2; ++sm) {
      const int r = wm * 32 + sm * 16 + l16;
      af[sm].q[0] = *(const uint4*)&As[cur][r][ak];
      af[sm].q[1] = *(const uint4*)&As[cur][r][ak + 8];
    }
#pragma unroll
    for (int sn = 0; sn < 4; ++sn) {
      const int c = wn * 64 + sn * 16 + l16;
      bfr[sn].q[0] = *(const uint4*)&Bs[cur][c][bk];
      bfr[sn].q[1] = *(const uint4*)&Bs[cur][c][bk + 8];
    }

#pragma unroll
    for (int sm = 0; sm < 2; ++sm)
#pragma unroll
      for (int sn = 0; sn < 4; ++sn)
        acc[sm][sn] = __builtin_amdgcn_wmma_f32_16x16x32_bf16(
            false, af[sm].v, false, bfr[sn].v, (short)0, acc[sm][sn],
            false, false);

    __syncthreads();   // orders: this tile fully read before it is re-staged
  }

  // ---- epilogue ----
#pragma unroll
  for (int sm = 0; sm < 2; ++sm) {
#pragma unroll
    for (int sn = 0; sn < 4; ++sn) {
      const int baseR = row0 + wm * 32 + sm * 16 + hi * 8;
      const int ccol  = col0 + wn * 64 + sn * 16 + l16;
#pragma unroll
      for (int r = 0; r < 8; ++r) {
        const int i = baseR + r;
        if (i >= Meff) continue;
        const float v = acc[sm][sn][r];
        switch (mode) {
          case EPI_STORE:
            if (C)   C[(size_t)i * ldc + ccol]   = v;
            if (Cbf) Cbf[(size_t)i * ldc + ccol] = f2bf(v);
            break;
          case EPI_SCATTER_SCALE_ACC: {
            const int tok = scatteridx[i];
            Cacc[(size_t)tok * ldc + ccol] +=
                v * comb[(size_t)tok * nexp + expert];
            break;
          }
          case EPI_GATE_ADD: {
            const float o = v * gatev[i] + addend[(size_t)i * ldc + ccol];
            if (Cbf) Cbf[(size_t)i * ldc + ccol] = f2bf(o);
            if (C)   C[(size_t)i * ldc + ccol]   = o;
            break;
          }
          case EPI_RESID:
            C[(size_t)i * ldc + ccol] = resid[(size_t)i * ldc + ccol] + v;
            break;
        }
      }
    }
  }
}

// ---------------------------------------------------------------------------
// RMSNorm: one 256-thread block per token; writes bf16 normalized row.
// ---------------------------------------------------------------------------
__global__ __launch_bounds__(256)
void rmsnorm_kernel(const float* __restrict__ h, const float* __restrict__ w,
                    u16* __restrict__ hn)
{
  const int n = blockIdx.x;
  const float* row = h + (size_t)n * D_;
  float ss = 0.f;
  for (int i = threadIdx.x; i < D_; i += 256) { const float x = row[i]; ss += x * x; }
  __shared__ float red[256];
  red[threadIdx.x] = ss;
  __syncthreads();
  for (int s = 128; s > 0; s >>= 1) {
    if (threadIdx.x < s) red[threadIdx.x] += red[threadIdx.x + s];
    __syncthreads();
  }
  const float inv = rsqrtf(red[0] * (1.f / D_) + 1e-6f);
  for (int i = threadIdx.x; i < D_; i += 256)
    hn[(size_t)n * D_ + i] = f2bf(row[i] * inv * w[i]);
}

// ---------------------------------------------------------------------------
// Router: one wave per token; lane == expert (E_ == 32 == wave32).
// ---------------------------------------------------------------------------
__global__ __launch_bounds__(256)
void router_kernel(const u16* __restrict__ hn, const float* __restrict__ rw,
                   const float* __restrict__ rb,
                   float* __restrict__ comb, int* __restrict__ cnt,
                   int* __restrict__ idxl, float* __restrict__ Psum)
{
  __shared__ float ps[32];
  const int tid = threadIdx.x, lane = tid & 31, wid = tid >> 5;
  if (tid < 32) ps[tid] = 0.f;
  __syncthreads();

  const int token = blockIdx.x * 8 + wid;
  const u16* row = hn + (size_t)token * D_;
  float accv = 0.f;
  for (int k = 0; k < D_; ++k)
    accv += bf2f(row[k]) * rw[(size_t)k * E_ + lane];

  const float score = 1.f / (1.f + __expf(-(accv + rb[lane])));
  atomicAdd(&ps[lane], score);

  float cur = score;
  float topv[4]; int topi[4];
#pragma unroll
  for (int t = 0; t < 4; ++t) {
    float v = cur; int id = lane;
    for (int off = 16; off; off >>= 1) {
      const float ov = __shfl_xor(v, off);
      const int   oi = __shfl_xor(id, off);
      if (ov > v || (ov == v && oi < id)) { v = ov; id = oi; }
    }
    topv[t] = v; topi[t] = id;
    if (lane == id) cur = -1e30f;
  }
  const float wsum = topv[0] + topv[1] + topv[2] + topv[3] + 1e-8f;
  float c = 0.f;
#pragma unroll
  for (int t = 0; t < 4; ++t) if (topi[t] == lane) c = topv[t] / wsum;
  comb[(size_t)token * E_ + lane] = c;

  if (lane < 4) {
    const int e = topi[lane];
    const int pos = atomicAdd(&cnt[e], 1);
    idxl[(size_t)e * N_ + pos] = token;
  }
  __syncthreads();
  if (tid < 32) atomicAdd(&Psum[tid], ps[tid]);
}

// ---------------------------------------------------------------------------
__global__ void lbloss_kernel(const int* __restrict__ cnt,
                              const float* __restrict__ Psum,
                              float* __restrict__ out)
{
  if (threadIdx.x == 0 && blockIdx.x == 0) {
    const float invN = 1.f / (float)N_;
    float s = 0.f;
    for (int e = 0; e < E_; ++e) s += ((float)cnt[e] * invN) * (Psum[e] * invN);
    *out = (float)E_ * s * AUX_;
  }
}

// GLU spike activation for routed experts (compact rows, device count guard)
__global__ __launch_bounds__(256)
void expert_act_kernel(const float* __restrict__ y, const float* __restrict__ vth,
                       u16* __restrict__ act, const int* __restrict__ cnt)
{
  const u32 gid = blockIdx.x * 256u + threadIdx.x;
  const int i = gid >> 10;           // / EH_ (1024)
  const int j = gid & (EH_ - 1);
  if (i >= *cnt) return;
  const float g  = y[(size_t)i * (2 * EH_) + j];
  const float l  = y[(size_t)i * (2 * EH_) + EH_ + j];
  const float vt = vth[j];
  const float s  = (g >= vt) ? vt : 0.f;
  act[(size_t)i * EH_ + j] = f2bf(s * l);
}

// GLU spike activation for the shared expert
__global__ __launch_bounds__(256)
void shared_act_kernel(const float* __restrict__ sy, const float* __restrict__ vth,
                       u16* __restrict__ sact)
{
  const size_t gid = (size_t)blockIdx.x * 256 + threadIdx.x;
  const size_t i = gid >> 11;        // / SH_ (2048)
  const int    j = (int)(gid & (SH_ - 1));
  const float g  = sy[i * (2 * SH_) + j];
  const float l  = sy[i * (2 * SH_) + SH_ + j];
  const float vt = vth[j];
  const float s  = (g >= vt) ? vt : 0.f;
  sact[i * SH_ + j] = f2bf(s * l);
}

// shared gate: one wave per token, dot(hn_row, gate_w) -> sigmoid
__global__ __launch_bounds__(256)
void gate_kernel(const u16* __restrict__ hn, const float* __restrict__ gw,
                 float* __restrict__ gate)
{
  const int tid = threadIdx.x, lane = tid & 31, wid = tid >> 5;
  const int token = blockIdx.x * 8 + wid;
  const u16* row = hn + (size_t)token * D_;
  float accv = 0.f;
  for (int k = lane; k < D_; k += 32) accv += bf2f(row[k]) * gw[k];
  for (int off = 16; off; off >>= 1) accv += __shfl_xor(accv, off);
  if (lane == 0) gate[token] = 1.f / (1.f + __expf(-accv));
}

// plain f32 -> bf16 convert
__global__ __launch_bounds__(256)
void f2bf_kernel(const float* __restrict__ src, u16* __restrict__ dst, size_t n)
{
  const size_t i = (size_t)blockIdx.x * 256 + threadIdx.x;
  if (i < n) dst[i] = f2bf(src[i]);
}

// f32 [K][N] -> bf16 transposed [N][K], LDS-tiled (32x32). blockIdx.z = slab.
__global__ __launch_bounds__(256)
void f2bf_transpose_kernel(const float* __restrict__ src, u16* __restrict__ dst,
                           int K, int Ncols)
{
  __shared__ float tile[32][33];
  const float* s = src + (size_t)blockIdx.z * K * Ncols;
  u16*         d = dst + (size_t)blockIdx.z * K * Ncols;
  const int k0 = blockIdx.x * 32, n0 = blockIdx.y * 32;
  const int tx = threadIdx.x & 31, ty = threadIdx.x >> 5;    // 32 x 8
  for (int r = ty; r < 32; r += 8)
    tile[r][tx] = s[(size_t)(k0 + r) * Ncols + n0 + tx];
  __syncthreads();
  for (int r = ty; r < 32; r += 8)
    d[(size_t)(n0 + r) * K + k0 + tx] = f2bf(tile[tx][r]);
}

// ---------------------------------------------------------------------------
static void launch_gemm(hipStream_t st,
                        const u16* A, int lda, const u16* BT, int ldbt,
                        int M, int Ncols, int K,
                        const int* rowidx, const int* cntp, const int* scat,
                        float* C, u16* Cbf, int ldc, int mode,
                        const float* comb, int expert, float* Cacc,
                        const float* gatev, const float* addend,
                        const float* resid)
{
  dim3 g((M + BM - 1) / BM, Ncols / BN), b(256);
  gemm_bf16_wmma<<<g, b, 0, st>>>(A, lda, BT, ldbt, M, Ncols, K,
                                  rowidx, cntp, scat, C, Cbf, ldc, mode,
                                  comb, expert, E_, Cacc, gatev, addend, resid);
}

static inline char* carve(char*& p, size_t bytes) {
  char* r = p;
  p += (bytes + 255) & ~(size_t)255;
  return r;
}

extern "C" void kernel_launch(void* const* d_in, const int* in_sizes, int n_in,
                              void* d_out, int out_size, void* d_ws, size_t ws_size,
                              hipStream_t stream)
{
  const float* h     = (const float*)d_in[0];
  const float* normw = (const float*)d_in[1];
  const float* w_ld  = (const float*)d_in[2];   // D x L
  const float* w_lu  = (const float*)d_in[3];   // L x D
  const float* w_r   = (const float*)d_in[4];   // D x E
  const float* b_r   = (const float*)d_in[5];   // E
  const float* w_e1  = (const float*)d_in[6];   // E x L x 2EH
  const float* vth_e = (const float*)d_in[7];   // E x EH
  const float* w_e2  = (const float*)d_in[8];   // E x EH x L
  const float* w_s1  = (const float*)d_in[9];   // D x 2SH
  const float* vth_s = (const float*)d_in[10];  // SH
  const float* w_s2  = (const float*)d_in[11];  // SH x D
  const float* w_g   = (const float*)d_in[12];  // D x 1
  const float* w_o   = (const float*)d_in[13];  // D x D
  float* out = (float*)d_out;                   // N*D floats + 1 loss scalar

  // ---- workspace carve (y aliases sy, act aliases sact) ----
  char* p = (char*)d_ws;
  u16*   hn     = (u16*)  carve(p, (size_t)N_ * D_ * 2);
  u16*   latent = (u16*)  carve(p, (size_t)N_ * L_ * 2);
  float* comb   = (float*)carve(p, (size_t)N_ * E_ * 4);
  int*   cnt    = (int*)  carve(p, (size_t)E_ * 4);
  float* Psum   = (float*)carve(p, (size_t)E_ * 4);
  int*   idxl   = (int*)  carve(p, (size_t)E_ * N_ * 4);
  float* eacc   = (float*)carve(p, (size_t)N_ * L_ * 4);
  u16*   eaccb  = (u16*)  carve(p, (size_t)N_ * L_ * 2);
  float* routed = (float*)carve(p, (size_t)N_ * D_ * 4);
  float* sy     = (float*)carve(p, (size_t)N_ * 2 * SH_ * 4);
  float* y      = sy;                          // expert phase finishes before shared
  u16*   sact   = (u16*)  carve(p, (size_t)N_ * SH_ * 2);
  u16*   act    = sact;
  float* gate   = (float*)carve(p, (size_t)N_ * 4);
  u16*   combo  = (u16*)  carve(p, (size_t)N_ * D_ * 2);
  // transposed bf16 weights: W[K][N] -> WT[N][K]
  u16*   wldT   = (u16*)  carve(p, (size_t)D_ * L_ * 2);
  u16*   wluT   = (u16*)  carve(p, (size_t)L_ * D_ * 2);
  u16*   we1T   = (u16*)  carve(p, (size_t)E_ * L_ * 2 * EH_ * 2);
  u16*   we2T   = (u16*)  carve(p, (size_t)E_ * EH_ * L_ * 2);
  u16*   ws1T   = (u16*)  carve(p, (size_t)D_ * 2 * SH_ * 2);
  u16*   ws2T   = (u16*)  carve(p, (size_t)SH_ * D_ * 2);
  u16*   woT    = (u16*)  carve(p, (size_t)D_ * D_ * 2);

  hipMemsetAsync(cnt,  0, (size_t)E_ * 4, stream);
  hipMemsetAsync(Psum, 0, (size_t)E_ * 4, stream);
  hipMemsetAsync(eacc, 0, (size_t)N_ * L_ * 4, stream);

  auto convT = [&](const float* s, u16* d, int K, int Ncols, int slabs) {
    f2bf_transpose_kernel<<<dim3(K / 32, Ncols / 32, slabs), dim3(256), 0,
                            stream>>>(s, d, K, Ncols);
  };
  convT(w_ld, wldT, D_,  L_,      1);
  convT(w_lu, wluT, L_,  D_,      1);
  convT(w_e1, we1T, L_,  2 * EH_, E_);
  convT(w_e2, we2T, EH_, L_,      E_);
  convT(w_s1, ws1T, D_,  2 * SH_, 1);
  convT(w_s2, ws2T, SH_, D_,      1);
  convT(w_o,  woT,  D_,  D_,      1);

  // 1) RMSNorm -> bf16
  rmsnorm_kernel<<<dim3(N_), dim3(256), 0, stream>>>(h, normw, hn);

  // 2) Router + top-4 + combine weights + aux-loss sums
  router_kernel<<<dim3(N_ / 8), dim3(256), 0, stream>>>(hn, w_r, b_r,
                                                        comb, cnt, idxl, Psum);
  lbloss_kernel<<<dim3(1), dim3(64), 0, stream>>>(cnt, Psum, out + (size_t)N_ * D_);

  // 3) latent = hn @ W_down  (store bf16 only)
  launch_gemm(stream, hn, D_, wldT, D_, N_, L_, D_,
              nullptr, nullptr, nullptr, nullptr, latent, L_, EPI_STORE,
              nullptr, 0, nullptr, nullptr, nullptr, nullptr);

  // 4) Routed experts: gather -> GEMM1 -> spike-GLU -> GEMM2 scatter-combine
  for (int e = 0; e < E_; ++e) {
    const int* elist = idxl + (size_t)e * N_;
    const int* ecnt  = cnt + e;
    launch_gemm(stream, latent, L_, we1T + (size_t)e * L_ * 2 * EH_, L_,
                N_, 2 * EH_, L_, elist, ecnt, nullptr,
                y, nullptr, 2 * EH_, EPI_STORE,
                nullptr, 0, nullptr, nullptr, nullptr, nullptr);
    expert_act_kernel<<<dim3((N_ * EH_) / 256), dim3(256), 0, stream>>>(
        y, vth_e + (size_t)e * EH_, act, ecnt);
    launch_gemm(stream, act, EH_, we2T + (size_t)e * EH_ * L_, EH_,
                N_, L_, EH_, nullptr, ecnt, elist,
                nullptr, nullptr, L_, EPI_SCATTER_SCALE_ACC,
                comb, e, eacc, nullptr, nullptr, nullptr);
  }

  // 5) routed = eacc @ W_up
  {
    const size_t n = (size_t)N_ * L_;
    f2bf_kernel<<<dim3((unsigned)((n + 255) / 256)), dim3(256), 0, stream>>>(
        eacc, eaccb, n);
  }
  launch_gemm(stream, eaccb, L_, wluT, L_, N_, D_, L_,
              nullptr, nullptr, nullptr, routed, nullptr, D_, EPI_STORE,
              nullptr, 0, nullptr, nullptr, nullptr, nullptr);

  // 6) Shared expert: sy = hn @ W_s1 ; spike-GLU ; gate
  launch_gemm(stream, hn, D_, ws1T, D_, N_, 2 * SH_, D_,
              nullptr, nullptr, nullptr, sy, nullptr, 2 * SH_, EPI_STORE,
              nullptr, 0, nullptr, nullptr, nullptr, nullptr);
  shared_act_kernel<<<dim3((unsigned)(((size_t)N_ * SH_) / 256)), dim3(256), 0,
                      stream>>>(sy, vth_s, sact);
  gate_kernel<<<dim3(N_ / 8), dim3(256), 0, stream>>>(hn, w_g, gate);

  // 7) combined = (sact @ W_s2) * gate + routed   (bf16 for next GEMM)
  launch_gemm(stream, sact, SH_, ws2T, SH_, N_, D_, SH_,
              nullptr, nullptr, nullptr, nullptr, combo, D_, EPI_GATE_ADD,
              nullptr, 0, nullptr, gate, routed, nullptr);

  // 8) out = h + combined @ W_out
  launch_gemm(stream, combo, D_, woT, D_, N_, D_, D_,
              nullptr, nullptr, nullptr, out, nullptr, D_, EPI_RESID,
              nullptr, 0, nullptr, nullptr, nullptr, h);
}
